// TransformerBlock_64536178590026
// MI455X (gfx1250) — matrix operand order repaired and checked by reading, not compiled, not run on hardware
//
#include <hip/hip_runtime.h>
#include <hip/hip_bf16.h>
#include <math.h>

#define DIMC 1024
#define HEADS 16
#define DHEAD 64
#define MLPD 4096
#define BBATCH 8
#define NSEQ 1024
#define MROWS (BBATCH*NSEQ)   /* 8192 */
#define EPSV 1e-5f

typedef __bf16 bf16_t;
typedef __attribute__((ext_vector_type(16))) __bf16 v16bf;
typedef __attribute__((ext_vector_type(8)))  __bf16 v8bf;
typedef __attribute__((ext_vector_type(8)))  float  v8f;
typedef int v4i_vs __attribute__((vector_size(16)));   // matches builtin param type

union V16 { v16bf v; v8bf h[2]; };

// ---------- CDNA5 async global->LDS copy (direct, tracked by ASYNCcnt) ----------
#if __has_builtin(__builtin_amdgcn_global_load_async_to_lds_b128) && \
    __has_builtin(__builtin_amdgcn_s_wait_asynccnt)
#define HAVE_ASYNC_LDS 1
#else
#define HAVE_ASYNC_LDS 0
#endif

__device__ inline void copy16_g2l(const bf16_t* __restrict__ gsrc, bf16_t* __restrict__ ldst) {
#if HAVE_ASYNC_LDS
  // generic->AS1/AS3 via integer round-trip (LDS generic addr low 32 bits == LDS offset)
  __attribute__((address_space(1))) v4i_vs* gs =
      (__attribute__((address_space(1))) v4i_vs*)(uintptr_t)gsrc;
  __attribute__((address_space(3))) v4i_vs* ls =
      (__attribute__((address_space(3))) v4i_vs*)(uint32_t)(uintptr_t)ldst;
  __builtin_amdgcn_global_load_async_to_lds_b128(gs, ls, 0, 0);
#else
  *reinterpret_cast<v8bf*>(ldst) = *reinterpret_cast<const v8bf*>(gsrc);
#endif
}
__device__ inline void copy_fence() {
#if HAVE_ASYNC_LDS
  __builtin_amdgcn_s_wait_asynccnt(0);
#endif
}

// ---------- small helpers ----------
__device__ inline float half16_max(float v) {
  #pragma unroll
  for (int m = 1; m < 16; m <<= 1) v = fmaxf(v, __shfl_xor(v, m, 32));
  return v;
}
__device__ inline float half16_sum(float v) {
  #pragma unroll
  for (int m = 1; m < 16; m <<= 1) v += __shfl_xor(v, m, 32);
  return v;
}
__device__ inline float wave32_sum(float v) {
  #pragma unroll
  for (int m = 1; m < 32; m <<= 1) v += __shfl_xor(v, m, 32);
  return v;
}
__device__ inline float block_reduce_sum(float v, float* sdat) {
  int tid = threadIdx.x;
  sdat[tid] = v; __syncthreads();
  for (int s = 128; s > 0; s >>= 1) { if (tid < s) sdat[tid] += sdat[tid + s]; __syncthreads(); }
  float r = sdat[0]; __syncthreads();
  return r;
}

// ---------- elementwise convert f32 -> bf16 ----------
__global__ __launch_bounds__(256) void k_f32_to_bf16(const float* __restrict__ in,
                                                     bf16_t* __restrict__ out, int n) {
  int i = (blockIdx.x * 256 + threadIdx.x) * 4;
  if (i + 3 < n) {
    float4 f = *reinterpret_cast<const float4*>(in + i);
    out[i + 0] = (bf16_t)f.x; out[i + 1] = (bf16_t)f.y;
    out[i + 2] = (bf16_t)f.z; out[i + 3] = (bf16_t)f.w;
  }
}

// ---------- tiled transpose-convert: f32 [K][N] -> bf16 [N][K] ----------
__global__ __launch_bounds__(256) void k_transpose_bf16(const float* __restrict__ in,
                                                        bf16_t* __restrict__ out,
                                                        int K, int N) {
  __shared__ float tile[32][33];
  int tx = threadIdx.x, ty = threadIdx.y;      // block (32,8)
  int n0 = blockIdx.x * 32, k0 = blockIdx.y * 32;
  #pragma unroll
  for (int i = 0; i < 4; i++) {
    int k = k0 + ty + i * 8;
    tile[ty + i * 8][tx] = in[(size_t)k * N + (n0 + tx)];
  }
  __syncthreads();
  #pragma unroll
  for (int i = 0; i < 4; i++) {
    int n = n0 + ty + i * 8;
    out[(size_t)n * K + (k0 + tx)] = (bf16_t)tile[tx][ty + i * 8];
  }
}

// ---------- modulated layernorm: xn = (LN(x)*g+b)*(1+scale)+shift, bf16 out ----------
__global__ __launch_bounds__(256) void k_modln(const float* __restrict__ x,
                                               const float* __restrict__ ss,
                                               const float* __restrict__ g,
                                               const float* __restrict__ bta,
                                               bf16_t* __restrict__ xn) {
  __shared__ float sdat[256];
  int row = blockIdx.x, tid = threadIdx.x;
  const float* xr = x + (size_t)row * DIMC;
  float v[4]; float s = 0.f, s2 = 0.f;
  #pragma unroll
  for (int i = 0; i < 4; i++) { v[i] = xr[tid + i * 256]; s += v[i]; s2 += v[i] * v[i]; }
  float st = block_reduce_sum(s, sdat);
  float s2t = block_reduce_sum(s2, sdat);
  float mean = st * (1.f / DIMC);
  float var = s2t * (1.f / DIMC) - mean * mean;
  float inv = rsqrtf(var + EPSV);
  const float* ssr = ss + (size_t)row * (2 * DIMC);
  bf16_t* xo = xn + (size_t)row * DIMC;
  #pragma unroll
  for (int i = 0; i < 4; i++) {
    int c = tid + i * 256;
    float xv = (v[i] - mean) * inv * g[c] + bta[c];
    xv = xv * (1.f + ssr[c]) + ssr[DIMC + c];
    xo[c] = (bf16_t)xv;
  }
}

// ---------- WMMA bf16 GEMM: C = A[MxK] * Bt[NxK]^T (+bias, +epilogue) ----------
enum { EPI_PLAIN = 0, EPI_RES = 1, EPI_QKVMLP = 2 };

__global__ __launch_bounds__(256) void k_gemm_bf16(
    const bf16_t* __restrict__ A, const bf16_t* __restrict__ Bt,
    const float* __restrict__ bias, const float* __restrict__ res,
    float* __restrict__ outF, bf16_t* __restrict__ outH,
    int M, int N, int K, int epi)
{
  __shared__ __align__(16) bf16_t As[128 * 40];
  __shared__ __align__(16) bf16_t Bs[128 * 40];
  const int tid = threadIdx.x, lane = tid & 31, wave = tid >> 5;
  const int wm = wave >> 2, wn = wave & 3;          // 2 x 4 wave grid
  const int khalf = lane >> 4, lrow = lane & 15;
  const size_t rowA0 = (size_t)blockIdx.y * 128;
  const size_t rowB0 = (size_t)blockIdx.x * 128;

  v8f acc[4][2];
  #pragma unroll
  for (int i = 0; i < 4; i++) { v8f z = {}; acc[i][0] = z; acc[i][1] = z; }

  for (int kt = 0; kt < K; kt += 32) {
    __syncthreads();
    #pragma unroll
    for (int p = 0; p < 2; p++) {
      int c = tid + p * 256;
      int r = c >> 2, c8 = (c & 3) * 8;
      copy16_g2l(A + (rowA0 + r) * K + kt + c8, As + r * 40 + c8);
      copy16_g2l(Bt + (rowB0 + r) * K + kt + c8, Bs + r * 40 + c8);
    }
    copy_fence();
    __syncthreads();

    V16 aF[4], bF[2];
    #pragma unroll
    for (int sm = 0; sm < 4; sm++) {
      const bf16_t* base = As + (wm * 64 + sm * 16 + lrow) * 40;
      aF[sm].h[0] = *reinterpret_cast<const v8bf*>(base + 8 * khalf);
      aF[sm].h[1] = *reinterpret_cast<const v8bf*>(base + 16 + 8 * khalf);
    }
    #pragma unroll
    for (int sn = 0; sn < 2; sn++) {
      const bf16_t* base = Bs + (wn * 32 + sn * 16 + lrow) * 40;
      bF[sn].h[0] = *reinterpret_cast<const v8bf*>(base + 8 * khalf);
      bF[sn].h[1] = *reinterpret_cast<const v8bf*>(base + 16 + 8 * khalf);
    }
    #pragma unroll
    for (int sm = 0; sm < 4; sm++)
      #pragma unroll
      for (int sn = 0; sn < 2; sn++)
        acc[sm][sn] = __builtin_amdgcn_wmma_f32_16x16x32_bf16(
            false, aF[sm].v, false, bF[sn].v, (short)0, acc[sm][sn], false, false);
  }

  // epilogue
  #pragma unroll
  for (int sm = 0; sm < 4; sm++)
    #pragma unroll
    for (int sn = 0; sn < 2; sn++)
      #pragma unroll
      for (int r = 0; r < 8; r++) {
        int row = blockIdx.y * 128 + wm * 64 + sm * 16 + khalf * 8 + r;
        int col = blockIdx.x * 128 + wn * 32 + sn * 16 + lrow;
        float v = acc[sm][sn][r] + (bias ? bias[col] : 0.f);
        if (epi == EPI_PLAIN) {
          outF[(size_t)row * N + col] = v;
        } else if (epi == EPI_RES) {
          outF[(size_t)row * N + col] = v + res[(size_t)row * N + col];
        } else {  // EPI_QKVMLP: cols<3072 -> f32 qkv; else silu -> bf16 mlp act
          if (col < 3 * DIMC) {
            outF[(size_t)row * (3 * DIMC) + col] = v;
          } else {
            float sig = 1.f / (1.f + __expf(-v));
            outH[(size_t)row * MLPD + (col - 3 * DIMC)] = (bf16_t)(v * sig);
          }
        }
      }
}

// ---------- per-head QK layernorm + repack ----------
// qkv f32 [B*N][3*DIM] -> qh,kh bf16 [B*H][N][64], vt bf16 [B*H][64][N]
__global__ __launch_bounds__(256) void k_qknorm(
    const float* __restrict__ qkv,
    const float* __restrict__ qg, const float* __restrict__ qb,
    const float* __restrict__ kg, const float* __restrict__ kb,
    bf16_t* __restrict__ qh, bf16_t* __restrict__ kh, bf16_t* __restrict__ vt)
{
  int w = blockIdx.x * 8 + (threadIdx.x >> 5);
  int lane = threadIdx.x & 31;
  int h = w & (HEADS - 1);
  int n = (w >> 4) & (NSEQ - 1);
  int b = w >> 14;
  size_t row = (size_t)(b * NSEQ + n);
  size_t bh = (size_t)(b * HEADS + h);
  int d0 = lane * 2;

  // q
  {
    float2 qv = reinterpret_cast<const float2*>(qkv + row * (3 * DIMC) + h * DHEAD)[lane];
    float s = wave32_sum(qv.x + qv.y);
    float s2 = wave32_sum(qv.x * qv.x + qv.y * qv.y);
    float mean = s * (1.f / DHEAD);
    float inv = rsqrtf(s2 * (1.f / DHEAD) - mean * mean + EPSV);
    bf16_t* o = qh + (bh * NSEQ + n) * DHEAD;
    o[d0]     = (bf16_t)((qv.x - mean) * inv * qg[d0] + qb[d0]);
    o[d0 + 1] = (bf16_t)((qv.y - mean) * inv * qg[d0 + 1] + qb[d0 + 1]);
  }
  // k
  {
    float2 kv = reinterpret_cast<const float2*>(qkv + row * (3 * DIMC) + DIMC + h * DHEAD)[lane];
    float s = wave32_sum(kv.x + kv.y);
    float s2 = wave32_sum(kv.x * kv.x + kv.y * kv.y);
    float mean = s * (1.f / DHEAD);
    float inv = rsqrtf(s2 * (1.f / DHEAD) - mean * mean + EPSV);
    bf16_t* o = kh + (bh * NSEQ + n) * DHEAD;
    o[d0]     = (bf16_t)((kv.x - mean) * inv * kg[d0] + kb[d0]);
    o[d0 + 1] = (bf16_t)((kv.y - mean) * inv * kg[d0 + 1] + kb[d0 + 1]);
  }
  // v (transposed: [d][n])
  {
    float2 vv = reinterpret_cast<const float2*>(qkv + row * (3 * DIMC) + 2 * DIMC + h * DHEAD)[lane];
    vt[(bh * DHEAD + d0) * NSEQ + n]     = (bf16_t)vv.x;
    vt[(bh * DHEAD + d0 + 1) * NSEQ + n] = (bf16_t)vv.y;
  }
}

// ---------- flash attention (per (b,h), 128-query block, 64-key steps) ----------
__global__ __launch_bounds__(256) void k_attn(
    const bf16_t* __restrict__ qh, const bf16_t* __restrict__ kh,
    const bf16_t* __restrict__ vt, bf16_t* __restrict__ o)
{
  __shared__ __align__(16) bf16_t Ks[64 * 72];
  __shared__ __align__(16) bf16_t Vs[64 * 72];
  __shared__ __align__(16) bf16_t Ps[8 * 16 * 72];
  const int tid = threadIdx.x, lane = tid & 31, wave = tid >> 5;
  const int khalf = lane >> 4, lrow = lane & 15;
  const int qb = blockIdx.x & 7;
  const size_t bh = blockIdx.x >> 3;

  const bf16_t* qbase = qh + bh * NSEQ * DHEAD;
  const bf16_t* kbase = kh + bh * NSEQ * DHEAD;
  const bf16_t* vbase = vt + bh * DHEAD * NSEQ;

  // load this wave's Q fragments (16 rows x 64 d), kept in registers
  V16 aQ[2];
  {
    int mrow = qb * 128 + wave * 16 + lrow;
    const bf16_t* qr = qbase + (size_t)mrow * DHEAD;
    #pragma unroll
    for (int c = 0; c < 2; c++) {
      aQ[c].h[0] = *reinterpret_cast<const v8bf*>(qr + c * 32 + 8 * khalf);
      aQ[c].h[1] = *reinterpret_cast<const v8bf*>(qr + c * 32 + 16 + 8 * khalf);
    }
  }

  float Mr[8], Lr[8];
  v8f Oacc[4];
  #pragma unroll
  for (int r = 0; r < 8; r++) { Mr[r] = -3.0e38f; Lr[r] = 0.f; }
  #pragma unroll
  for (int d = 0; d < 4; d++) { v8f z = {}; Oacc[d] = z; }

  bf16_t* Pw = Ps + wave * 16 * 72;

  for (int j = 0; j < NSEQ; j += 64) {
    __syncthreads();
    #pragma unroll
    for (int p = 0; p < 2; p++) {
      int c = tid + p * 256;
      int r = c >> 3, c8 = (c & 7) * 8;
      copy16_g2l(kbase + (size_t)(j + r) * DHEAD + c8, Ks + r * 72 + c8);
      copy16_g2l(vbase + (size_t)r * NSEQ + j + c8, Vs + r * 72 + c8);
    }
    copy_fence();
    __syncthreads();

    // S = Q * K^T (16 x 64), 4 tiles, K-dim 64 via 2 chained WMMAs
    v8f S[4];
    #pragma unroll
    for (int t = 0; t < 4; t++) { v8f z = {}; S[t] = z; }
    #pragma unroll
    for (int t = 0; t < 4; t++) {
      const bf16_t* base = Ks + (t * 16 + lrow) * 72;
      #pragma unroll
      for (int c = 0; c < 2; c++) {
        V16 bK;
        bK.h[0] = *reinterpret_cast<const v8bf*>(base + c * 32 + 8 * khalf);
        bK.h[1] = *reinterpret_cast<const v8bf*>(base + c * 32 + 16 + 8 * khalf);
        S[t] = __builtin_amdgcn_wmma_f32_16x16x32_bf16(
            false, aQ[c].v, false, bK.v, (short)0, S[t], false, false);
      }
    }

    // online softmax over this wave's 16 rows (row data lives in 16-lane halves)
    #pragma unroll
    for (int r = 0; r < 8; r++) {
      float mx = -3.0e38f;
      #pragma unroll
      for (int t = 0; t < 4; t++) { S[t][r] *= 0.125f; mx = fmaxf(mx, S[t][r]); }
      mx = half16_max(mx);
      float mnew = fmaxf(Mr[r], mx);
      float corr = __expf(Mr[r] - mnew);
      Mr[r] = mnew;
      float ls = 0.f;
      #pragma unroll
      for (int t = 0; t < 4; t++) { float p = __expf(S[t][r] - mnew); S[t][r] = p; ls += p; }
      ls = half16_sum(ls);
      Lr[r] = Lr[r] * corr + ls;
      #pragma unroll
      for (int d = 0; d < 4; d++) Oacc[d][r] *= corr;
    }

    // stage P (C-layout -> A-frag layout via per-wave LDS)
    #pragma unroll
    for (int t = 0; t < 4; t++)
      #pragma unroll
      for (int r = 0; r < 8; r++)
        Pw[(khalf * 8 + r) * 72 + t * 16 + lrow] = (bf16_t)S[t][r];

    // O += P * V   (K-dim = 64 keys via 2 chains, N = 64 d via 4 tiles)
    #pragma unroll
    for (int dn = 0; dn < 4; dn++) {
      #pragma unroll
      for (int c = 0; c < 2; c++) {
        V16 aP, bV;
        const bf16_t* pb = Pw + lrow * 72 + c * 32;
        aP.h[0] = *reinterpret_cast<const v8bf*>(pb + 8 * khalf);
        aP.h[1] = *reinterpret_cast<const v8bf*>(pb + 16 + 8 * khalf);
        const bf16_t* vb = Vs + (dn * 16 + lrow) * 72 + c * 32;
        bV.h[0] = *reinterpret_cast<const v8bf*>(vb + 8 * khalf);
        bV.h[1] = *reinterpret_cast<const v8bf*>(vb + 16 + 8 * khalf);
        Oacc[dn] = __builtin_amdgcn_wmma_f32_16x16x32_bf16(
            false, aP.v, false, bV.v, (short)0, Oacc[dn], false, false);
      }
    }
  }

  // normalize and store o as bf16 [b][n][h*64+d]
  size_t b = bh >> 4;
  int h = (int)(bh & 15);
  #pragma unroll
  for (int dn = 0; dn < 4; dn++)
    #pragma unroll
    for (int r = 0; r < 8; r++) {
      int nq = qb * 128 + wave * 16 + khalf * 8 + r;
      float val = Oacc[dn][r] / Lr[r];
      o[(b * NSEQ + nq) * DIMC + h * DHEAD + dn * 16 + lrow] = (bf16_t)val;
    }
}

// ================= host launch =================
extern "C" void kernel_launch(void* const* d_in, const int* in_sizes, int n_in,
                              void* d_out, int out_size, void* d_ws, size_t ws_size,
                              hipStream_t stream) {
  (void)in_sizes; (void)n_in; (void)out_size; (void)ws_size;
  const float* x       = (const float*)d_in[0];
  const float* emb     = (const float*)d_in[1];
  const float* W_emb   = (const float*)d_in[2];
  const float* b_emb   = (const float*)d_in[3];
  const float* ln_g    = (const float*)d_in[4];
  const float* ln_b    = (const float*)d_in[5];
  const float* W_fused = (const float*)d_in[6];
  const float* b_fused = (const float*)d_in[7];
  const float* qn_g    = (const float*)d_in[8];
  const float* qn_b    = (const float*)d_in[9];
  const float* kn_g    = (const float*)d_in[10];
  const float* kn_b    = (const float*)d_in[11];
  const float* W_ao    = (const float*)d_in[12];
  const float* b_ao    = (const float*)d_in[13];
  const float* W_mlp   = (const float*)d_in[14];
  const float* b_mlp   = (const float*)d_in[15];
  float* out = (float*)d_out;

  char* ws = (char*)d_ws;
  size_t off = 0;
  auto alloc = [&](size_t bytes) { size_t o = off; off += (bytes + 255) & ~(size_t)255; return o; };

  bf16_t* Wt_emb   = (bf16_t*)(ws + alloc((size_t)2048 * 1024 * 2));
  bf16_t* Wt_fused = (bf16_t*)(ws + alloc((size_t)7168 * 1024 * 2));
  bf16_t* Wt_ao    = (bf16_t*)(ws + alloc((size_t)1024 * 1024 * 2));
  bf16_t* Wt_mlp   = (bf16_t*)(ws + alloc((size_t)1024 * 4096 * 2));
  bf16_t* emb16    = (bf16_t*)(ws + alloc((size_t)MROWS * DIMC * 2));
  float*  ss       = (float*) (ws + alloc((size_t)MROWS * 2 * DIMC * 4));
  bf16_t* xn       = (bf16_t*)(ws + alloc((size_t)MROWS * DIMC * 2));
  float*  qkv      = (float*) (ws + alloc((size_t)MROWS * 3 * DIMC * 4));
  bf16_t* mlp_act  = (bf16_t*)(ws + alloc((size_t)MROWS * MLPD * 2));
  bf16_t* qh       = (bf16_t*)(ws + alloc((size_t)MROWS * DIMC * 2));
  bf16_t* kh       = (bf16_t*)(ws + alloc((size_t)MROWS * DIMC * 2));
  bf16_t* vth      = (bf16_t*)(ws + alloc((size_t)MROWS * DIMC * 2));
  bf16_t* obuf     = (bf16_t*)(ws + alloc((size_t)MROWS * DIMC * 2));
  float*  x1       = (float*) (ws + alloc((size_t)MROWS * DIMC * 4));

  dim3 tb(256);
  // 1) converts / transposes
  k_f32_to_bf16<<<(MROWS * DIMC) / 1024, tb, 0, stream>>>(emb, emb16, MROWS * DIMC);
  {
    dim3 bt(32, 8);
    k_transpose_bf16<<<dim3(2048 / 32, 1024 / 32), bt, 0, stream>>>(W_emb, Wt_emb, 1024, 2048);
    k_transpose_bf16<<<dim3(7168 / 32, 1024 / 32), bt, 0, stream>>>(W_fused, Wt_fused, 1024, 7168);
    k_transpose_bf16<<<dim3(1024 / 32, 1024 / 32), bt, 0, stream>>>(W_ao, Wt_ao, 1024, 1024);
    k_transpose_bf16<<<dim3(1024 / 32, 4096 / 32), bt, 0, stream>>>(W_mlp, Wt_mlp, 4096, 1024);
  }
  // 2) ss = emb @ W_emb + b_emb
  k_gemm_bf16<<<dim3(2048 / 128, MROWS / 128), tb, 0, stream>>>(
      emb16, Wt_emb, b_emb, nullptr, ss, nullptr, MROWS, 2048, 1024, EPI_PLAIN);
  // 3) modulated LN
  k_modln<<<MROWS, tb, 0, stream>>>(x, ss, ln_g, ln_b, xn);
  // 4) fused = xn @ W_fused + b_fused -> qkv f32 + silu(mlp) bf16
  k_gemm_bf16<<<dim3(7168 / 128, MROWS / 128), tb, 0, stream>>>(
      xn, Wt_fused, b_fused, nullptr, qkv, mlp_act, MROWS, 7168, 1024, EPI_QKVMLP);
  // 5) qk-norm + repack
  k_qknorm<<<(BBATCH * NSEQ * HEADS) / 8, tb, 0, stream>>>(
      qkv, qn_g, qn_b, kn_g, kn_b, qh, kh, vth);
  // 6) attention
  k_attn<<<BBATCH * HEADS * (NSEQ / 128), tb, 0, stream>>>(qh, kh, vth, obuf);
  // 7) x1 = x + o @ W_ao + b_ao
  k_gemm_bf16<<<dim3(1024 / 128, MROWS / 128), tb, 0, stream>>>(
      obuf, Wt_ao, b_ao, x, x1, nullptr, MROWS, 1024, 1024, EPI_RES);
  // 8) out = x1 + mlp_act @ W_mlp + b_mlp
  k_gemm_bf16<<<dim3(1024 / 128, MROWS / 128), tb, 0, stream>>>(
      mlp_act, Wt_mlp, b_mlp, x1, out, nullptr, MROWS, 1024, 4096, EPI_RES);
}
